// SNN_14422500180578
// MI455X (gfx1250) — matrix operand order repaired and checked by reading, not compiled, not run on hardware
//
#include <hip/hip_runtime.h>

typedef __attribute__((ext_vector_type(16))) _Float16 v16h;
typedef __attribute__((ext_vector_type(8)))  float    v8f;

#define TAU_LEAK 0.95f   /* 1 - DT/TAU */
#define V_TH     1.0f

union Frag16 { v16h v; uint32_t u[8]; };

// ---- ISA per-lane fragment loaders (16-bit A 16x32, B 32x16) ----
// A: lane L<16 -> row M=L, K in {0..7, 16..23}; lane L>=16 -> row M=L-16, K in {8..15, 24..31}
// element pair j holds halves (2j, 2j+1):  k = (j>=4 ? 16 : 0) + 8*hi + 2*(j&3)
// -> per lane: two contiguous 16B chunks (compiler emits 2x global_load_b128)
__device__ __forceinline__ v16h load_a_frag(const _Float16* __restrict__ S,
                                            int row, int ldk, int kk, int hi) {
  Frag16 a;
  const uint32_t* base = (const uint32_t*)(S + (size_t)row * ldk + kk);
#pragma unroll
  for (int j = 0; j < 8; ++j) {
    int koff = ((j & 4) ? 16 : 0) + 8 * hi + 2 * (j & 3);
    a.u[j] = base[koff >> 1];
  }
  return a.v;
}

// B: lane L -> col N=L&15; element pair j: k = (j>=4 ? 16 : 0) + 8*hi + 2*(j&3)  (two 16B chunks, same as A)
__device__ __forceinline__ v16h load_b_frag(const _Float16* __restrict__ W,
                                            int ncol, int ldk, int kk, int hi) {
  Frag16 b;
  const uint32_t* base = (const uint32_t*)(W + (size_t)ncol * ldk + kk);
#pragma unroll
  for (int j = 0; j < 8; ++j) {
    int koff = ((j & 4) ? 16 : 0) + 8 * hi + 2 * (j & 3);
    b.u[j] = base[koff >> 1];
  }
  return b.v;
}

// ---- spike train generation: s = (u < clip((x+1)*0.02, 0, 1)) ----
__global__ __launch_bounds__(256) void spike_gen_kernel(
    const float* __restrict__ x, const float* __restrict__ u_t,
    _Float16* __restrict__ s0, int n) {
  int i = blockIdx.x * blockDim.x + threadIdx.x;
  if (i < n) {
    float p = (x[i] + 1.0f) * 0.02f;
    p = fminf(fmaxf(p, 0.0f), 1.0f);
    s0[i] = (u_t[i] < p) ? (_Float16)1.0f : (_Float16)0.0f;
  }
}

__global__ __launch_bounds__(256) void f32_to_f16_kernel(
    const float* __restrict__ src, _Float16* __restrict__ dst, int n) {
  int i = blockIdx.x * blockDim.x + threadIdx.x;
  if (i < n) dst[i] = (_Float16)src[i];
}

// ---- fused GEMM (I = S @ W^T + b) + LIF update ----
// Block = 8 waves covering 128 rows; all waves share ONE 64-col N-strip so their
// B (weight) loads are identical addresses -> served by WGP$ after first wave.
// Each wave owns a 16x64 strip: 1 A-frag + 4 B-frags + 4 WMMAs per k-step.
// unroll 2: gives the scheduler two independent load groups so iteration i+1's
// loads can issue above iteration i's WMMAs (no register-rotation copies).
__global__ __launch_bounds__(256) void gemm_lif_kernel(
    const _Float16* __restrict__ S,   // [B, K] spikes (f16 {0,1})
    const _Float16* __restrict__ Wh,  // [N, K] weights f16
    const float*    __restrict__ bias,// [N]
    float*          __restrict__ V,   // [B, N] membrane potential (persistent)
    _Float16*       __restrict__ Sout,// [B, N] output spikes f16
    int K, int N) {
  const int lane = threadIdx.x & 31;
  const int wave = threadIdx.x >> 5;          // 0..7
  const int hi   = lane >> 4;
  const int mr   = lane & 15;
  const int m0   = blockIdx.y * 128 + wave * 16;  // waves split M
  const int n0   = blockIdx.x * 64;               // block shares N-strip
  const int arow = m0 + mr;

  v8f acc[4] = {v8f{}, v8f{}, v8f{}, v8f{}};

#pragma unroll 2
  for (int kk = 0; kk < K; kk += 32) {
    __builtin_prefetch(S + (size_t)arow * K + kk + 64, 0, 3); // speculative; OOB dropped
    v16h a = load_a_frag(S, arow, K, kk, hi);
    v16h b[4];
#pragma unroll
    for (int s = 0; s < 4; ++s)
      b[s] = load_b_frag(Wh, n0 + s * 16 + mr, K, kk, hi);
#pragma unroll
    for (int s = 0; s < 4; ++s)
      acc[s] = __builtin_amdgcn_wmma_f32_16x16x32_f16(
          /*neg_a=*/false, a, /*neg_b=*/false, b[s],
          /*c_mod=*/(short)0, acc[s], /*reuse_a=*/false, /*reuse_b=*/false);
  }

  // LIF epilogue: v = 0.95*v + I; s = (v > V_TH); v = s ? 0 : v
  // C/D layout: VGPR r -> M = r + 8*hi, N = lane&15
#pragma unroll
  for (int s = 0; s < 4; ++s) {
    const int n  = n0 + s * 16 + mr;
    const float bn = bias[n];
    size_t idx = (size_t)(m0 + 8 * hi) * N + n;   // step by N per row: no per-elem mul
#pragma unroll
    for (int r = 0; r < 8; ++r, idx += N) {
      float I = acc[s][r] + bn;
      float v = V[idx] * TAU_LEAK + I;
      bool sp = v > V_TH;
      Sout[idx] = sp ? (_Float16)1.0f : (_Float16)0.0f;
      V[idx] = sp ? 0.0f : v;
    }
  }
}

// ---- final layer: N padded to 16 (weight rows 10..15 zero); accumulates spikes into out ----
__global__ __launch_bounds__(32) void gemm_lif_out_kernel(
    const _Float16* __restrict__ S,    // [B, K]
    const _Float16* __restrict__ Wh,   // [16, K] padded f16
    const float*    __restrict__ bias, // [10]
    float*          __restrict__ V,    // [B, 16] padded membrane
    float*          __restrict__ Out,  // [B, 10] spike-count accumulator
    int K, int NOUT) {
  const int lane = threadIdx.x & 31;
  const int hi = lane >> 4;
  const int mr = lane & 15;
  const int m0 = blockIdx.x * 16;

  v8f acc = {};
#pragma unroll 2
  for (int kk = 0; kk < K; kk += 32) {
    v16h a = load_a_frag(S, m0 + mr, K, kk, hi);
    v16h b = load_b_frag(Wh, mr, K, kk, hi);
    acc = __builtin_amdgcn_wmma_f32_16x16x32_f16(
        false, a, false, b, (short)0, acc, false, false);
  }

  const int n = mr;
  const float bn = (n < NOUT) ? bias[n] : 0.0f;
#pragma unroll
  for (int r = 0; r < 8; ++r) {
    const int m = m0 + r + 8 * hi;
    const int idx = m * 16 + n;
    float I = acc[r] + bn;
    float v = V[idx] * TAU_LEAK + I;
    bool sp = v > V_TH;
    V[idx] = sp ? 0.0f : v;
    if (n < NOUT && sp) Out[m * NOUT + n] += 1.0f; // (m,n) owned by one lane; stream-ordered
  }
}

extern "C" void kernel_launch(void* const* d_in, const int* in_sizes, int n_in,
                              void* d_out, int out_size, void* d_ws, size_t ws_size,
                              hipStream_t stream) {
  constexpr int T = 50, B = 256, IN = 2048, HID = 4096, NOUT = 10;
  const float* x  = (const float*)d_in[0];
  const float* u  = (const float*)d_in[1];
  const float* W0 = (const float*)d_in[2];
  const float* b0 = (const float*)d_in[3];
  const float* W1 = (const float*)d_in[4];
  const float* b1 = (const float*)d_in[5];
  const float* W2 = (const float*)d_in[6];
  const float* b2 = (const float*)d_in[7];
  float* out = (float*)d_out;

  char* ws = (char*)d_ws;
  size_t off = 0;
  auto carve = [&](size_t bytes) -> void* {
    void* p = ws + off;
    off = (off + bytes + 255) & ~(size_t)255;
    return p;
  };
  _Float16* W0h = (_Float16*)carve((size_t)HID * IN * 2);   // 16 MB
  _Float16* W1h = (_Float16*)carve((size_t)HID * HID * 2);  // 32 MB
  _Float16* W2h = (_Float16*)carve((size_t)16 * HID * 2);   // padded [16, HID]
  _Float16* s0  = (_Float16*)carve((size_t)B * IN * 2);
  _Float16* s1  = (_Float16*)carve((size_t)B * HID * 2);
  _Float16* s2  = (_Float16*)carve((size_t)B * HID * 2);
  float*    v0  = (float*)carve((size_t)B * HID * 4);
  float*    v1  = (float*)carve((size_t)B * HID * 4);
  float*    v2  = (float*)carve((size_t)B * 16 * 4);

  // zero persistent state / padded weights / output accumulator (capture-safe)
  hipMemsetAsync(v0, 0, (size_t)B * HID * 4, stream);
  hipMemsetAsync(v1, 0, (size_t)B * HID * 4, stream);
  hipMemsetAsync(v2, 0, (size_t)B * 16 * 4, stream);
  hipMemsetAsync(W2h, 0, (size_t)16 * HID * 2, stream);
  hipMemsetAsync(out, 0, (size_t)B * NOUT * 4, stream);

  // convert weights to f16 once per call
  {
    int n = HID * IN;
    f32_to_f16_kernel<<<(n + 255) / 256, 256, 0, stream>>>(W0, W0h, n);
    n = HID * HID;
    f32_to_f16_kernel<<<(n + 255) / 256, 256, 0, stream>>>(W1, W1h, n);
    n = NOUT * HID; // fills rows 0..9 of padded W2h (rows 10..15 stay zero)
    f32_to_f16_kernel<<<(n + 255) / 256, 256, 0, stream>>>(W2, W2h, n);
  }

  const dim3 blk(256);                    // 8 waves
  const dim3 grid_gemm(HID / 64, B / 128); // (64, 2)

  for (int t = 0; t < T; ++t) {
    const int nSpk = B * IN;
    spike_gen_kernel<<<(nSpk + 255) / 256, 256, 0, stream>>>(
        x, u + (size_t)t * B * IN, s0, nSpk);

    gemm_lif_kernel<<<grid_gemm, blk, 0, stream>>>(s0, W0h, b0, v0, s1, IN, HID);
    gemm_lif_kernel<<<grid_gemm, blk, 0, stream>>>(s1, W1h, b1, v1, s2, HID, HID);
    gemm_lif_out_kernel<<<B / 16, 32, 0, stream>>>(s2, W2h, b2, v2, out, HID, NOUT);
  }
}
